// DepthToGaussian_8744553414986
// MI455X (gfx1250) — matrix operand order repaired and checked by reading, not compiled, not run on hardware
//
#include <hip/hip_runtime.h>
#include <math.h>

// ---------------------------------------------------------------------------
// DepthToGaussian for MI455X (gfx1250): bf16 WMMA implicit-GEMM conv stack
// with compile-time-shaped address math, fragment-ready swizzled weights
// (L2-resident, no B LDS stage), two-pass batchnorm (conv bias absorbed by
// BN -> skipped, exact), register-resident FPS, head 1x1 convs evaluated only
// on the 2048 FPS-selected rows per batch (pointwise => exact).
// ---------------------------------------------------------------------------

typedef __bf16 bf16;
typedef __attribute__((ext_vector_type(16))) __bf16 v16bf;
typedef __attribute__((ext_vector_type(8)))  float  v8f;

__device__ __forceinline__ float gelu_exact(float x) {
  return 0.5f * x * (1.f + erff(x * 0.70710678118654752f));
}
__device__ __forceinline__ float softplus_f(float x) {
  return log1pf(expf(-fabsf(x))) + fmaxf(x, 0.f);
}
__device__ __forceinline__ float sigmoid_f(float x) {
  return 1.f / (1.f + expf(-x));
}

// --------------------------- input packing --------------------------------
// rgb (B,3,512,512) f32 + depth (B,1,512,512) f32  ->  NHWC bf16 (B,512,512,4)
__global__ void __launch_bounds__(256) pack_input_kernel(
    const float* __restrict__ rgb, const float* __restrict__ depth,
    bf16* __restrict__ X) {
  size_t i = (size_t)blockIdx.x * 256 + threadIdx.x;          // over B*512*512
  if (i >= (size_t)8 * 512 * 512) return;
  int b = (int)(i >> 18);
  int p = (int)(i & 262143);
  const size_t HW = 262144;
  X[i * 4 + 0] = (bf16)rgb[(size_t)b * 3 * HW + 0 * HW + p];
  X[i * 4 + 1] = (bf16)rgb[(size_t)b * 3 * HW + 1 * HW + p];
  X[i * 4 + 2] = (bf16)rgb[(size_t)b * 3 * HW + 2 * HW + p];
  X[i * 4 + 3] = (bf16)depth[(size_t)b * HW + p];
}

// --------------------------- weight repack (fragment-swizzled) ------------
// OIHW f32 -> bf16 in WMMA-B-fragment-ready order:
//   Wsw[ ((k>>5)*128 + o)*32 + (k&31) ] = W[k][o],  k = (kh*KW+kw)*Cin + cin
// Each lane's B fragment (16 consecutive k of one column) is then a single
// contiguous, 32B-aligned global load. Zero padded in K (to Kpad) and N.
__global__ void __launch_bounds__(256) prep_weight_kernel(
    const float* __restrict__ w, bf16* __restrict__ out,
    int O, int I, int KH, int KW, int Kpad) {
  int idx = blockIdx.x * 256 + threadIdx.x;                   // over Kpad*128
  if (idx >= Kpad * 128) return;
  int k = idx >> 7, o = idx & 127;
  bf16 v = (bf16)0.f;
  int Kreal = I * KH * KW;
  if (k < Kreal && o < O) {
    int cin = k % I;
    int f = k / I;
    int kh = f / KW, kw = f - kh * KW;
    v = (bf16)w[(((size_t)o * I + cin) * KH + kh) * KW + kw];
  }
  out[(((size_t)(k >> 5) * 128 + o) << 5) + (k & 31)] = v;
}

// --------------------------- WMMA fragment helpers ------------------------
// 16-bit A 16x32 layout (matched to B): lanes 0-15 = rows M0..15, K 0..15 in
// V0..7; lanes 16-31 = same rows, K 16..31.
__device__ __forceinline__ v16bf load_a_frag(const bf16 (*As)[40], int wave, int lane) {
  int hi = lane >> 4, ln = lane & 15;
  int mrow = wave * 16 + ln;
  v16bf a;
#pragma unroll
  for (int v = 0; v < 8; ++v) {
    int kb = 2 * v + hi * 16;
    a[2 * v]     = As[mrow][kb];
    a[2 * v + 1] = As[mrow][kb + 1];
  }
  return a;
}
// B 32x16 fragment: lanes 0-15 hold K=0..15 of column N=lane, lanes 16-31 hold
// K=16..31. With the swizzled weight layout this is one 32B contiguous load.
__device__ __forceinline__ v16bf load_b_frag_global(const bf16* __restrict__ Wsw,
                                                    int ks, int col, int hi) {
  return *reinterpret_cast<const v16bf*>(Wsw + (((size_t)ks * 128 + col) << 5) + hi * 16);
}

// --------------------------- implicit-GEMM conv ---------------------------
// Y[m, cout] = sum_k A[m,k] * W[k,cout], m = (b,oy,ox), k = (kh,kw,cin).
// Tile: 128 (M) x 128 (N=Cout), 256 threads = 8 waves, each wave a 16x128
// strip via 8 accumulators of v_wmma_f32_16x16x32_bf16.  All shape math is
// compile time (shifts), A staged via b128 vector loads, B read straight from
// L2-resident swizzled weights.  Grid is exact: B*Hout*Wout % 128 == 0.
template <int Cin, int KH, int KW, int STRIDE, int PAD,
          int Hin, int Win, int Hout, int Wout>
__global__ void __launch_bounds__(256) conv_wmma_t(
    const bf16* __restrict__ X, const bf16* __restrict__ Wsw, bf16* __restrict__ Y) {
  constexpr int Kreal = Cin * KH * KW;
  constexpr int Kpad = (Kreal + 31) & ~31;
  constexpr int KSTEPS = Kpad / 32;
  constexpr int HW = Hout * Wout;                 // power of two
  __shared__ bf16 As[128][40];                    // 80B rows: 16B-aligned
  const int lane = threadIdx.x & 31;
  const int wave = threadIdx.x >> 5;
  const int hi = lane >> 4;
  const int ln = lane & 15;
  const long long m0 = (long long)blockIdx.x * 128;

  v8f acc[8];
#pragma unroll
  for (int nt = 0; nt < 8; ++nt) acc[nt] = (v8f){0.f, 0.f, 0.f, 0.f, 0.f, 0.f, 0.f, 0.f};

  for (int ks = 0; ks < KSTEPS; ++ks) {
    const int k0 = ks * 32;
    __syncthreads();
    if constexpr (Cin == 128) {
      // one K-chunk = 32 consecutive input channels of one filter tap:
      // contiguous 64B in NHWC memory -> uint4 loads.
      const int f = k0 >> 7;
      const int kh = f / KW, kw = f - (f / KW) * KW;
      const int cin0 = k0 & 127;
      for (int t = threadIdx.x; t < 128 * 4; t += 256) {
        const int r = t >> 2, q = (t & 3) << 3;   // 8 halves per uint4
        const long long m = m0 + r;
        const int bidx = (int)(m / HW);
        const int rem = (int)(m - (long long)bidx * HW);
        const int oy = rem / Wout, ox = rem - (rem / Wout) * Wout;
        const int iy = oy * STRIDE - PAD + kh;
        const int ix = ox * STRIDE - PAD + kw;
        uint4 val = {0u, 0u, 0u, 0u};
        if (iy >= 0 && iy < Hin && ix >= 0 && ix < Win)
          val = *reinterpret_cast<const uint4*>(
              X + (((size_t)bidx * Hin + iy) * Win + ix) * 128 + cin0 + q);
        *reinterpret_cast<uint4*>(&As[r][q]) = val;
      }
    } else {
      // small Cin (4): aligned u32 pair loads (pairs never straddle a tap).
      for (int t = threadIdx.x; t < 128 * 16; t += 256) {
        const int r = t >> 4, kk = (t & 15) << 1;
        const int k = k0 + kk;
        unsigned int val = 0u;
        if (k < Kreal) {
          const int cin = k & (Cin - 1);
          const int f = k / Cin;
          const int kh = f / KW, kw = f - (f / KW) * KW;
          const long long m = m0 + r;
          const int bidx = (int)(m / HW);
          const int rem = (int)(m - (long long)bidx * HW);
          const int oy = rem / Wout, ox = rem - (rem / Wout) * Wout;
          const int iy = oy * STRIDE - PAD + kh;
          const int ix = ox * STRIDE - PAD + kw;
          if (iy >= 0 && iy < Hin && ix >= 0 && ix < Win)
            val = *reinterpret_cast<const unsigned int*>(
                X + (((size_t)bidx * Hin + iy) * Win + ix) * Cin + cin);
        }
        *reinterpret_cast<unsigned int*>(&As[r][kk]) = val;
      }
    }
    __syncthreads();

    v16bf a = load_a_frag(As, wave, lane);
#pragma unroll
    for (int nt = 0; nt < 8; ++nt) {
      v16bf b = load_b_frag_global(Wsw, ks, nt * 16 + ln, hi);
      acc[nt] = __builtin_amdgcn_wmma_f32_16x16x32_bf16(
          false, a, false, b, (short)0, acc[nt], false, false);
    }
  }

#pragma unroll
  for (int nt = 0; nt < 8; ++nt) {
    int col = nt * 16 + ln;
#pragma unroll
    for (int r = 0; r < 8; ++r) {
      long long m = m0 + wave * 16 + hi * 8 + r;
      Y[m * 128 + col] = (bf16)acc[nt][r];
    }
  }
}

// --------------------------- batchnorm ------------------------------------
__global__ void zero_stats_kernel(float* s) {
  if (threadIdx.x < 256) s[threadIdx.x] = 0.f;
}

// per-channel sum / sumsq over M rows of a [M x 128] bf16 activation
__global__ void __launch_bounds__(256) bn_stats_kernel(
    const bf16* __restrict__ y, float* __restrict__ stats, long long M) {
  __shared__ float s1[128], s2[128];
  if (threadIdx.x < 128) { s1[threadIdx.x] = 0.f; s2[threadIdx.x] = 0.f; }
  __syncthreads();
  int c = threadIdx.x & 127;
  int rl = threadIdx.x >> 7;   // 0..1
  float a1 = 0.f, a2 = 0.f;
  for (long long r = (long long)blockIdx.x * 2 + rl; r < M; r += (long long)gridDim.x * 2) {
    float v = (float)y[r * 128 + c];
    a1 += v; a2 += v * v;
  }
  atomicAdd(&s1[c], a1);
  atomicAdd(&s2[c], a2);
  __syncthreads();
  if (threadIdx.x < 128) {
    atomicAdd(&stats[threadIdx.x], s1[threadIdx.x]);
    atomicAdd(&stats[128 + threadIdx.x], s2[threadIdx.x]);
  }
}

// out = gelu( bn(yin)*g + be  [+ resid] )   (bf16 in / bf16 out)
__global__ void __launch_bounds__(256) bn_apply_kernel(
    const bf16* __restrict__ yin, bf16* __restrict__ out,
    const float* __restrict__ stats, const float* __restrict__ g,
    const float* __restrict__ be, const bf16* __restrict__ resid,
    long long total, float invM) {
  long long i = (long long)blockIdx.x * 256 + threadIdx.x;
  if (i >= total) return;
  int c = (int)(i & 127);
  float mean = stats[c] * invM;
  float var = fmaxf(stats[128 + c] * invM - mean * mean, 0.f);
  float v = ((float)yin[i] - mean) * rsqrtf(var + 1e-5f) * g[c] + be[c];
  if (resid) v += (float)resid[i];
  out[i] = (bf16)gelu_exact(v);
}

// --------------------------- backproject + 4x4 mean pool ------------------
__global__ void __launch_bounds__(256) pool_points_kernel(
    const float* __restrict__ depth, const float* __restrict__ K,
    float* __restrict__ pts) {
  int i = blockIdx.x * 256 + threadIdx.x;                     // over 8*128*128
  if (i >= 8 * 16384) return;
  int b = i >> 14;
  int gy = (i >> 7) & 127;
  int gx = i & 127;
  float fx = K[b * 9 + 0], cx = K[b * 9 + 2];
  float fy = K[b * 9 + 4], cy = K[b * 9 + 5];
  float sx = 0.f, sy = 0.f, sz = 0.f;
#pragma unroll
  for (int dv = 0; dv < 4; ++dv)
#pragma unroll
    for (int du = 0; du < 4; ++du) {
      int v = gy * 4 + dv, u = gx * 4 + du;
      float d = depth[((size_t)b * 512 + v) * 512 + u];
      sx += ((float)u - cx) * d / fx;
      sy += ((float)v - cy) * d / fy;
      sz += d;
    }
  pts[(size_t)i * 3 + 0] = sx * 0.0625f;
  pts[(size_t)i * 3 + 1] = sy * 0.0625f;
  pts[(size_t)i * 3 + 2] = sz * 0.0625f;
}

// --------------------------- farthest point sampling ----------------------
// One 1024-thread workgroup per batch. Each thread owns 16 points + running
// distances in registers; block argmax via LDS tree reduce. 2048 serial steps.
__global__ void __launch_bounds__(1024) fps_kernel(
    const float* __restrict__ pts, int* __restrict__ gidx) {
  __shared__ float rv[1024];
  __shared__ int ri[1024];
  const int b = blockIdx.x;
  const int tid = threadIdx.x;
  const float* P = pts + (size_t)b * 16384 * 3;
  float lx[16], ly[16], lz[16], ld[16];
#pragma unroll
  for (int j = 0; j < 16; ++j) {
    int i = tid + j * 1024;
    lx[j] = P[(size_t)i * 3 + 0];
    ly[j] = P[(size_t)i * 3 + 1];
    lz[j] = P[(size_t)i * 3 + 2];
    ld[j] = 3.4e38f;
  }
  int far = 0;
  for (int s = 0; s < 2048; ++s) {
    if (tid == 0) gidx[b * 2048 + s] = far;   // scan emits incoming farthest
    float cx = P[(size_t)far * 3 + 0];
    float cy = P[(size_t)far * 3 + 1];
    float cz = P[(size_t)far * 3 + 2];
    float best = -1.f;
    int bi = 0;
#pragma unroll
    for (int j = 0; j < 16; ++j) {
      float dx = lx[j] - cx, dy = ly[j] - cy, dz = lz[j] - cz;
      float d = dx * dx + dy * dy + dz * dz;
      float nd = fminf(ld[j], d);
      ld[j] = nd;
      if (nd > best) { best = nd; bi = tid + j * 1024; }
    }
    rv[tid] = best;
    ri[tid] = bi;
    __syncthreads();
    for (int offr = 512; offr > 0; offr >>= 1) {
      if (tid < offr) {
        if (rv[tid + offr] > rv[tid]) { rv[tid] = rv[tid + offr]; ri[tid] = ri[tid + offr]; }
      }
      __syncthreads();
    }
    far = ri[0];
    __syncthreads();
  }
}

// --------------------------- head GEMM (gathered rows) --------------------
// 128x128 tile, K=128 (4 fully-unrolled WMMA steps). A rows either dense
// (Adense) or gathered from feat via gidx. Fused bias (+optional GELU).
// Writes bf16 (outB, ld 128) and/or f32 (outF, ld outFld). Grid exact: M=16384.
__global__ void __launch_bounds__(256) gemm_head_kernel(
    const bf16* __restrict__ Adense, const bf16* __restrict__ feat,
    const int* __restrict__ gidx, const bf16* __restrict__ Wsw,
    const float* __restrict__ bias, int nbias,
    bf16* __restrict__ outB, float* __restrict__ outF, int outFld,
    int doGelu) {
  __shared__ bf16 As[128][40];
  const int lane = threadIdx.x & 31;
  const int wave = threadIdx.x >> 5;
  const int hi = lane >> 4;
  const int ln = lane & 15;
  const int m0 = blockIdx.x * 128;

  v8f acc[8];
#pragma unroll
  for (int nt = 0; nt < 8; ++nt) acc[nt] = (v8f){0.f, 0.f, 0.f, 0.f, 0.f, 0.f, 0.f, 0.f};

#pragma unroll
  for (int ks = 0; ks < 4; ++ks) {
    const int k0 = ks * 32;
    __syncthreads();
    for (int t = threadIdx.x; t < 512; t += 256) {
      const int r = t >> 2, q = (t & 3) << 3;
      const int m = m0 + r;
      const bf16* row;
      if (gidx) {
        int b = m >> 11;                 // 2048 selected per batch
        int sp = gidx[m];
        row = feat + (((size_t)b << 14) + sp) * 128;
      } else {
        row = Adense + (size_t)m * 128;
      }
      *reinterpret_cast<uint4*>(&As[r][q]) =
          *reinterpret_cast<const uint4*>(row + k0 + q);
    }
    __syncthreads();

    v16bf a = load_a_frag(As, wave, lane);
#pragma unroll
    for (int nt = 0; nt < 8; ++nt) {
      v16bf b = load_b_frag_global(Wsw, ks, nt * 16 + ln, hi);
      acc[nt] = __builtin_amdgcn_wmma_f32_16x16x32_bf16(
          false, a, false, b, (short)0, acc[nt], false, false);
    }
  }

#pragma unroll
  for (int nt = 0; nt < 8; ++nt) {
    int col = nt * 16 + ln;
#pragma unroll
    for (int r = 0; r < 8; ++r) {
      int m = m0 + wave * 16 + hi * 8 + r;
      float v = acc[nt][r] + ((col < nbias) ? bias[col] : 0.f);
      if (doGelu) v = gelu_exact(v);
      if (outB) outB[(size_t)m * 128 + col] = (bf16)v;
      if (outF && col < outFld) outF[(size_t)m * outFld + col] = v;
    }
  }
}

// --------------------------- final assembly -------------------------------
// out row (38): pt(3) | softplus scale(3) | unit quat(4) | sigmoid op(1) | sh(27)
__global__ void __launch_bounds__(256) assemble_kernel(
    const float* __restrict__ pts, const int* __restrict__ gidx,
    const float* __restrict__ prsel, float* __restrict__ out) {
  int m = blockIdx.x * 256 + threadIdx.x;                     // over 8*2048
  if (m >= 8 * 2048) return;
  int b = m >> 11;
  int sp = gidx[m];
  const float* P = pts + ((size_t)b * 16384 + sp) * 3;
  const float* pr = prsel + (size_t)m * 48;
  float* o = out + (size_t)m * 38;
  o[0] = P[0]; o[1] = P[1]; o[2] = P[2];
#pragma unroll
  for (int j = 0; j < 3; ++j) o[3 + j] = softplus_f(pr[j]);
  float r0 = pr[3], r1 = pr[4], r2 = pr[5], r3 = pr[6];
  float n = fmaxf(sqrtf(r0 * r0 + r1 * r1 + r2 * r2 + r3 * r3), 1e-12f);
  o[6] = r0 / n; o[7] = r1 / n; o[8] = r2 / n; o[9] = r3 / n;
  o[10] = sigmoid_f(pr[7]);
#pragma unroll
  for (int j = 0; j < 27; ++j) o[11 + j] = pr[8 + j];
}

// ===========================================================================
extern "C" void kernel_launch(void* const* d_in, const int* in_sizes, int n_in,
                              void* d_out, int out_size, void* d_ws, size_t ws_size,
                              hipStream_t stream) {
  (void)in_sizes; (void)n_in; (void)out_size; (void)ws_size;

  // Input flattening assumed to follow JAX pytree order (dict keys sorted):
  // 0:depth 1:intrinsics 2:b0 3:b1 4:be0 5:be1 6:g0 7:g1 8:hb0 9:hb1
  // 10:hw0 11:hw1 12..43: res[i]{b1,b2,be1,be2,g1,g2,w1,w2} 44:w0 45:w1 46:rgb
  const float* depth = (const float*)d_in[0];
  const float* intr  = (const float*)d_in[1];
  const float* g0    = (const float*)d_in[6];
  const float* be0   = (const float*)d_in[4];
  const float* g1    = (const float*)d_in[7];
  const float* be1   = (const float*)d_in[5];
  const float* hb0   = (const float*)d_in[8];
  const float* hb1   = (const float*)d_in[9];
  const float* hw0   = (const float*)d_in[10];
  const float* hw1   = (const float*)d_in[11];
  const float* w0    = (const float*)d_in[44];
  const float* w1    = (const float*)d_in[45];
  const float* rgb   = (const float*)d_in[46];
  // conv biases b0/b1/res-b* are absorbed by batchnorm -> unused (exact).

  char* base = (char*)d_ws;
  size_t off = 0;
  auto alloc = [&](size_t bytes) -> void* {
    off = (off + 255) & ~(size_t)255;
    void* p = base + off;
    off += bytes;
    return p;
  };

  bf16* Xin  = (bf16*)alloc((size_t)8 * 512 * 512 * 4 * 2);     // 16.8 MB
  bf16* W0p  = (bf16*)alloc((size_t)224 * 128 * 2);
  bf16* Wc[9];
  for (int i = 0; i < 9; ++i) Wc[i] = (bf16*)alloc((size_t)1152 * 128 * 2);
  bf16* Wh0  = (bf16*)alloc((size_t)128 * 128 * 2);
  bf16* Wh1  = (bf16*)alloc((size_t)128 * 128 * 2);
  bf16* act0 = (bf16*)alloc((size_t)524288 * 128 * 2);          // 134 MB
  bf16* act1 = (bf16*)alloc((size_t)131072 * 128 * 2);          // 33.5 MB (x)
  bf16* act2 = (bf16*)alloc((size_t)131072 * 128 * 2);          // h / raw
  bf16* act3 = (bf16*)alloc((size_t)131072 * 128 * 2);          // raw
  float* stats = (float*)alloc(256 * 4);
  float* pts  = (float*)alloc((size_t)8 * 16384 * 3 * 4);
  int*  gidx  = (int*)alloc((size_t)8 * 2048 * 4);
  bf16* tsel  = (bf16*)alloc((size_t)16384 * 128 * 2);
  float* prsel = (float*)alloc((size_t)16384 * 48 * 4);

  // --- pack input, repack weights into fragment-swizzled bf16 ---
  pack_input_kernel<<<8192, 256, 0, stream>>>(rgb, depth, Xin);
  prep_weight_kernel<<<(224 * 128 + 255) / 256, 256, 0, stream>>>(w0, W0p, 128, 4, 7, 7, 224);
  prep_weight_kernel<<<(1152 * 128 + 255) / 256, 256, 0, stream>>>(w1, Wc[0], 128, 128, 3, 3, 1152);
  for (int i = 0; i < 4; ++i) {
    const float* rw1 = (const float*)d_in[12 + 8 * i + 6];
    const float* rw2 = (const float*)d_in[12 + 8 * i + 7];
    prep_weight_kernel<<<(1152 * 128 + 255) / 256, 256, 0, stream>>>(rw1, Wc[1 + 2 * i], 128, 128, 3, 3, 1152);
    prep_weight_kernel<<<(1152 * 128 + 255) / 256, 256, 0, stream>>>(rw2, Wc[2 + 2 * i], 128, 128, 3, 3, 1152);
  }
  prep_weight_kernel<<<(128 * 128 + 255) / 256, 256, 0, stream>>>(hw0, Wh0, 128, 128, 1, 1, 128);
  prep_weight_kernel<<<(128 * 128 + 255) / 256, 256, 0, stream>>>(hw1, Wh1, 35, 128, 1, 1, 128);

  const long long M0 = 524288, M1 = 131072;

  // --- stem conv0: 7x7 s2 p3, 4->128, out 256x256 ---
  conv_wmma_t<4, 7, 7, 2, 3, 512, 512, 256, 256><<<4096, 256, 0, stream>>>(Xin, W0p, act0);
  zero_stats_kernel<<<1, 256, 0, stream>>>(stats);
  bn_stats_kernel<<<1024, 256, 0, stream>>>(act0, stats, M0);
  bn_apply_kernel<<<(unsigned)(M0 * 128 / 256), 256, 0, stream>>>(act0, act0, stats, g0, be0, nullptr, M0 * 128, 1.f / (float)M0);

  // --- stem conv1: 3x3 s2 p1, 128->128, out 128x128 -> x in act1 ---
  conv_wmma_t<128, 3, 3, 2, 1, 256, 256, 128, 128><<<1024, 256, 0, stream>>>(act0, Wc[0], act2);
  zero_stats_kernel<<<1, 256, 0, stream>>>(stats);
  bn_stats_kernel<<<1024, 256, 0, stream>>>(act2, stats, M1);
  bn_apply_kernel<<<(unsigned)(M1 * 128 / 256), 256, 0, stream>>>(act2, act1, stats, g1, be1, nullptr, M1 * 128, 1.f / (float)M1);

  // --- 4 residual blocks ---
  for (int i = 0; i < 4; ++i) {
    const float* rg1  = (const float*)d_in[12 + 8 * i + 4];
    const float* rbe1 = (const float*)d_in[12 + 8 * i + 2];
    const float* rg2  = (const float*)d_in[12 + 8 * i + 5];
    const float* rbe2 = (const float*)d_in[12 + 8 * i + 3];
    conv_wmma_t<128, 3, 3, 1, 1, 128, 128, 128, 128><<<1024, 256, 0, stream>>>(act1, Wc[1 + 2 * i], act2);
    zero_stats_kernel<<<1, 256, 0, stream>>>(stats);
    bn_stats_kernel<<<1024, 256, 0, stream>>>(act2, stats, M1);
    bn_apply_kernel<<<(unsigned)(M1 * 128 / 256), 256, 0, stream>>>(act2, act2, stats, rg1, rbe1, nullptr, M1 * 128, 1.f / (float)M1);
    conv_wmma_t<128, 3, 3, 1, 1, 128, 128, 128, 128><<<1024, 256, 0, stream>>>(act2, Wc[2 + 2 * i], act3);
    zero_stats_kernel<<<1, 256, 0, stream>>>(stats);
    bn_stats_kernel<<<1024, 256, 0, stream>>>(act3, stats, M1);
    bn_apply_kernel<<<(unsigned)(M1 * 128 / 256), 256, 0, stream>>>(act3, act1, stats, rg2, rbe2, act1, M1 * 128, 1.f / (float)M1);
  }

  // --- geometry: pooled backprojection + FPS ---
  pool_points_kernel<<<512, 256, 0, stream>>>(depth, intr, pts);
  fps_kernel<<<8, 1024, 0, stream>>>(pts, gidx);

  // --- head (only on the 8*2048 selected rows; pointwise => exact) ---
  gemm_head_kernel<<<128, 256, 0, stream>>>(nullptr, act1, gidx, Wh0, hb0, 128, tsel, nullptr, 0, 1);
  gemm_head_kernel<<<128, 256, 0, stream>>>(tsel, nullptr, nullptr, Wh1, hb1, 35, nullptr, prsel, 48, 0);

  // --- assemble gaussians ---
  assemble_kernel<<<64, 256, 0, stream>>>(pts, gidx, prsel, (float*)d_out);
}
